// CrossGCN_38482906972407
// MI455X (gfx1250) — matrix operand order repaired and checked by reading, not compiled, not run on hardware
//
#include <hip/hip_runtime.h>
#include <hip/hip_bf16.h>

// ---------------------------------------------------------------------------
// Problem constants (from reference)
// ---------------------------------------------------------------------------
enum : int {
    Bb = 8, Nn = 256, NI = 36, Ss = 64,
    IN_D = 1024, QUE_D = 1024, IMG_D = 2048, SEM_D = 512,
    P = 512, GATE = 512, CAT = 3 * GATE, OUT_D = 1024,
    ROWS = Bb * Nn            // 2048
};

typedef __attribute__((ext_vector_type(16))) __bf16 v16bf;
typedef __attribute__((ext_vector_type(8)))  float  v8f;

union FragBF { v16bf v; uint4 u[2]; };

// ---------------------------------------------------------------------------
// Generic bf16 WMMA GEMM:  C[M x N] = A[M x K] * Bt[N x K]^T  (+bias, epilogue)
// Block: 256 thr = 8 waves, 2 (M) x 4 (N); wave tile 16 x 64; K-step 32.
// Requires: M % 32 == 0, N % 256 == 0, K % 32 == 0 (true for all launches).
// EPI 0: f32 store (acc+bias)      EPI 1: bf16 store (acc+bias)
// EPI 2: bf16 store sigmoid(acc+bias) * gsrc   (gated fusion)
// ---------------------------------------------------------------------------
template<int EPI>
__global__ __launch_bounds__(256) void gemm_bf16_wmma(
    const __bf16* __restrict__ A,  int lda,
    const __bf16* __restrict__ Bt, int ldb,
    const float*  __restrict__ bias,
    float*        __restrict__ Cf,
    __bf16*       __restrict__ Cb, int ldc,
    const __bf16* __restrict__ gsrc,
    int K)
{
    const int lane  = threadIdx.x & 31;
    const int wave  = threadIdx.x >> 5;
    const int wm    = wave & 1;
    const int wn    = wave >> 1;
    const int tileM = blockIdx.y * 32  + wm * 16;
    const int tileN = blockIdx.x * 256 + wn * 64;

    const int lidx  = lane & 15;
    const int lhalf = lane >> 4;
    const int segA0 = 16 * lhalf;          // A: elems 0..7  = K + 16h .. +7
    const int segA1 = 16 * lhalf + 8;      // A: elems 8..15 = K + 16h + 8 .. +15
    const int segB  = 16 * lhalf;          // B: elems 0..15 = K + 16h .. +15

    const __bf16* ap  = A  + (size_t)(tileM + lidx) * lda;
    const __bf16* bp0 = Bt + (size_t)(tileN +  0 + lidx) * ldb + segB;
    const __bf16* bp1 = Bt + (size_t)(tileN + 16 + lidx) * ldb + segB;
    const __bf16* bp2 = Bt + (size_t)(tileN + 32 + lidx) * ldb + segB;
    const __bf16* bp3 = Bt + (size_t)(tileN + 48 + lidx) * ldb + segB;

    v8f acc0 = {}, acc1 = {}, acc2 = {}, acc3 = {};

    for (int k0 = 0; k0 < K; k0 += 32) {
        FragBF a, b0, b1, b2, b3;
        a.u[0]  = *(const uint4*)(ap  + k0 + segA0);
        a.u[1]  = *(const uint4*)(ap  + k0 + segA1);
        b0.u[0] = *(const uint4*)(bp0 + k0);
        b0.u[1] = *(const uint4*)(bp0 + k0 + 8);
        b1.u[0] = *(const uint4*)(bp1 + k0);
        b1.u[1] = *(const uint4*)(bp1 + k0 + 8);
        b2.u[0] = *(const uint4*)(bp2 + k0);
        b2.u[1] = *(const uint4*)(bp2 + k0 + 8);
        b3.u[0] = *(const uint4*)(bp3 + k0);
        b3.u[1] = *(const uint4*)(bp3 + k0 + 8);

        if (k0 + 64 <= K) {            // speculative prefetch of next K tile
            __builtin_prefetch(ap  + k0 + 32, 0, 0);
            __builtin_prefetch(bp0 + k0 + 32, 0, 0);
        }

        acc0 = __builtin_amdgcn_wmma_f32_16x16x32_bf16(false, a.v, false, b0.v,
                                                       (short)0, acc0, false, false);
        acc1 = __builtin_amdgcn_wmma_f32_16x16x32_bf16(false, a.v, false, b1.v,
                                                       (short)0, acc1, false, false);
        acc2 = __builtin_amdgcn_wmma_f32_16x16x32_bf16(false, a.v, false, b2.v,
                                                       (short)0, acc2, false, false);
        acc3 = __builtin_amdgcn_wmma_f32_16x16x32_bf16(false, a.v, false, b3.v,
                                                       (short)0, acc3, false, false);
    }

    const int c0 = tileN + lidx, c1 = c0 + 16, c2 = c0 + 32, c3 = c0 + 48;
    const float bi0 = bias[c0], bi1 = bias[c1], bi2 = bias[c2], bi3 = bias[c3];

#pragma unroll
    for (int r = 0; r < 8; ++r) {
        const int row = tileM + 8 * lhalf + r;
        const size_t base = (size_t)row * ldc;
        float v0 = acc0[r] + bi0;
        float v1 = acc1[r] + bi1;
        float v2 = acc2[r] + bi2;
        float v3 = acc3[r] + bi3;
        if constexpr (EPI == 0) {
            Cf[base + c0] = v0; Cf[base + c1] = v1;
            Cf[base + c2] = v2; Cf[base + c3] = v3;
        } else if constexpr (EPI == 1) {
            Cb[base + c0] = (__bf16)v0; Cb[base + c1] = (__bf16)v1;
            Cb[base + c2] = (__bf16)v2; Cb[base + c3] = (__bf16)v3;
        } else {
            float g0 = 1.0f / (1.0f + __expf(-v0));
            float g1 = 1.0f / (1.0f + __expf(-v1));
            float g2 = 1.0f / (1.0f + __expf(-v2));
            float g3 = 1.0f / (1.0f + __expf(-v3));
            Cb[base + c0] = (__bf16)(g0 * (float)gsrc[base + c0]);
            Cb[base + c1] = (__bf16)(g1 * (float)gsrc[base + c1]);
            Cb[base + c2] = (__bf16)(g2 * (float)gsrc[base + c2]);
            Cb[base + c3] = (__bf16)(g3 * (float)gsrc[base + c3]);
        }
    }
}

// ---------------------------------------------------------------------------
// node_feat = concat(h, que) in bf16  [ROWS x 2048]
// ---------------------------------------------------------------------------
__global__ __launch_bounds__(256) void k_node_feat(
    const float* __restrict__ h, const float* __restrict__ que,
    __bf16* __restrict__ nf)
{
    int idx = blockIdx.x * 256 + threadIdx.x;       // over ROWS*2048
    int row = idx >> 11;
    int c   = idx & 2047;
    int b   = row >> 8;
    float v = (c < IN_D) ? h[(size_t)row * IN_D + c]
                         : que[(size_t)b * QUE_D + (c - IN_D)];
    nf[idx] = (__bf16)v;
}

// f32 [K x N] -> bf16 [N x K] tiled transpose (K, N multiples of 32)
__global__ __launch_bounds__(256) void k_trans_bf16(
    const float* __restrict__ W, __bf16* __restrict__ Wt, int K, int N)
{
    __shared__ float tile[32][33];
    int tx = threadIdx.x & 31;
    int ty = threadIdx.x >> 5;      // 0..7
#pragma unroll
    for (int j = 0; j < 4; ++j) {
        int k = blockIdx.y * 32 + ty + j * 8;
        tile[ty + j * 8][tx] = W[(size_t)k * N + blockIdx.x * 32 + tx];
    }
    __syncthreads();
#pragma unroll
    for (int j = 0; j < 4; ++j) {
        int n = blockIdx.x * 32 + ty + j * 8;
        Wt[(size_t)n * K + blockIdx.y * 32 + tx] = (__bf16)tile[tx][ty + j * 8];
    }
}

__global__ __launch_bounds__(256) void k_cvt_bf16(
    const float* __restrict__ x, __bf16* __restrict__ y, int n)
{
    int i = blockIdx.x * 256 + threadIdx.x;
    if (i < n) y[i] = (__bf16)x[i];
}

// ---------------------------------------------------------------------------
// Additive attention over image keys: one block per (b,n) row.
// logits[k] = sum_p tanh(np[p] + ip[b,k,p]) * w[p] + b; softmax over k;
// ctx[d] = sum_k att[k] * img[b,k,d]   (bf16 out for the next WMMA GEMM)
// ---------------------------------------------------------------------------
__global__ __launch_bounds__(256) void k_attn_img(
    const float* __restrict__ npf, const float* __restrict__ ipf,
    const float* __restrict__ wv_, const float* __restrict__ bsc,
    const float* __restrict__ img, __bf16* __restrict__ ctx)
{
    __shared__ float s_att[40];
    const int row = blockIdx.x;
    const int b   = row >> 8;
    const int t    = threadIdx.x;
    const int wave = t >> 5, lane = t & 31;

    float npv[16], wv[16];
#pragma unroll
    for (int i = 0; i < 16; ++i) {
        npv[i] = npf[(size_t)row * P + lane + 32 * i];
        wv[i]  = wv_[lane + 32 * i];
    }
    for (int k = wave; k < NI; k += 8) {
        const float* ipr = ipf + ((size_t)b * NI + k) * P;
        float part = 0.f;
#pragma unroll
        for (int i = 0; i < 16; ++i)
            part += tanhf(npv[i] + ipr[lane + 32 * i]) * wv[i];
#pragma unroll
        for (int off = 16; off > 0; off >>= 1)
            part += __shfl_down(part, off);
        if (lane == 0) s_att[k] = part + bsc[0];
    }
    __syncthreads();
    if (t == 0) {
        float mx = -1e30f;
        for (int k = 0; k < NI; ++k) mx = fmaxf(mx, s_att[k]);
        float den = 0.f, e[NI];
        for (int k = 0; k < NI; ++k) { e[k] = __expf(s_att[k] - mx); den += e[k]; }
        float inv = 1.f / den;
        for (int k = 0; k < NI; ++k) s_att[k] = e[k] * inv;
    }
    __syncthreads();
#pragma unroll
    for (int j = 0; j < 8; ++j) {
        int d = t + j * 256;
        float s = 0.f;
        for (int k = 0; k < NI; ++k)
            s += s_att[k] * img[((size_t)b * NI + k) * IMG_D + d];
        ctx[(size_t)row * IMG_D + d] = (__bf16)s;
    }
}

// Same over semantic keys (S=64, D=512)
__global__ __launch_bounds__(256) void k_attn_sem(
    const float* __restrict__ npf, const float* __restrict__ spf,
    const float* __restrict__ wv_, const float* __restrict__ bsc,
    const float* __restrict__ sem, __bf16* __restrict__ ctx)
{
    __shared__ float s_att[64];
    const int row = blockIdx.x;
    const int b   = row >> 8;
    const int t    = threadIdx.x;
    const int wave = t >> 5, lane = t & 31;

    float npv[16], wv[16];
#pragma unroll
    for (int i = 0; i < 16; ++i) {
        npv[i] = npf[(size_t)row * P + lane + 32 * i];
        wv[i]  = wv_[lane + 32 * i];
    }
    for (int k = wave; k < Ss; k += 8) {
        const float* spr = spf + ((size_t)b * Ss + k) * P;
        float part = 0.f;
#pragma unroll
        for (int i = 0; i < 16; ++i)
            part += tanhf(npv[i] + spr[lane + 32 * i]) * wv[i];
#pragma unroll
        for (int off = 16; off > 0; off >>= 1)
            part += __shfl_down(part, off);
        if (lane == 0) s_att[k] = part + bsc[0];
    }
    __syncthreads();
    if (t == 0) {
        float mx = -1e30f;
        for (int k = 0; k < Ss; ++k) mx = fmaxf(mx, s_att[k]);
        float den = 0.f, e[Ss];
        for (int k = 0; k < Ss; ++k) { e[k] = __expf(s_att[k] - mx); den += e[k]; }
        float inv = 1.f / den;
        for (int k = 0; k < Ss; ++k) s_att[k] = e[k] * inv;
    }
    __syncthreads();
#pragma unroll
    for (int j = 0; j < 2; ++j) {
        int d = t + j * 256;
        float s = 0.f;
        for (int k = 0; k < Ss; ++k)
            s += s_att[k] * sem[((size_t)b * Ss + k) * SEM_D + d];
        ctx[(size_t)row * SEM_D + d] = (__bf16)s;
    }
}

// ---------------------------------------------------------------------------
// Host orchestration
// ---------------------------------------------------------------------------
static void launch_gemm(int epi, const __bf16* A, int lda, const __bf16* Bt, int ldb,
                        const float* bias, float* Cf, __bf16* Cb, int ldc,
                        const __bf16* gsrc, int M, int N, int K, hipStream_t s)
{
    dim3 g(N / 256, M / 32), b(256);
    if (epi == 0)      gemm_bf16_wmma<0><<<g, b, 0, s>>>(A, lda, Bt, ldb, bias, Cf, Cb, ldc, gsrc, K);
    else if (epi == 1) gemm_bf16_wmma<1><<<g, b, 0, s>>>(A, lda, Bt, ldb, bias, Cf, Cb, ldc, gsrc, K);
    else               gemm_bf16_wmma<2><<<g, b, 0, s>>>(A, lda, Bt, ldb, bias, Cf, Cb, ldc, gsrc, K);
}

extern "C" void kernel_launch(void* const* d_in, const int* in_sizes, int n_in,
                              void* d_out, int out_size, void* d_ws, size_t ws_size,
                              hipStream_t stream)
{
    const float* h      = (const float*)d_in[0];
    const float* img    = (const float*)d_in[1];
    const float* sem    = (const float*)d_in[2];
    const float* que    = (const float*)d_in[3];
    const float* W_cif  = (const float*)d_in[4];  const float* b_cif  = (const float*)d_in[5];
    const float* W_cii  = (const float*)d_in[6];  const float* b_cii  = (const float*)d_in[7];
    const float* w_ia   = (const float*)d_in[8];  const float* b_ia   = (const float*)d_in[9];
    const float* W_csf  = (const float*)d_in[10]; const float* b_csf  = (const float*)d_in[11];
    const float* W_csn  = (const float*)d_in[12]; const float* b_csn  = (const float*)d_in[13];
    const float* w_sa   = (const float*)d_in[14]; const float* b_sa   = (const float*)d_in[15];
    const float* W_ig   = (const float*)d_in[16]; const float* b_ig   = (const float*)d_in[17];
    const float* W_sg   = (const float*)d_in[18]; const float* b_sg   = (const float*)d_in[19];
    const float* W_fg   = (const float*)d_in[20]; const float* b_fg   = (const float*)d_in[21];
    const float* W_gate = (const float*)d_in[22]; const float* b_gate = (const float*)d_in[23];
    const float* W_out  = (const float*)d_in[24]; const float* b_out  = (const float*)d_in[25];
    float* out = (float*)d_out;

    char* ws = (char*)d_ws;
    size_t off = 0;
    auto alloc = [&](size_t bytes) -> void* {
        void* p = ws + off;
        off += (bytes + 255) & ~(size_t)255;
        return p;
    };

    const int NF_K = IN_D + QUE_D;   // 2048
    __bf16* nf      = (__bf16*)alloc((size_t)ROWS * NF_K * 2);
    __bf16* wt_cif  = (__bf16*)alloc((size_t)P * NF_K * 2);
    __bf16* wt_cii  = (__bf16*)alloc((size_t)P * IMG_D * 2);
    __bf16* wt_csf  = (__bf16*)alloc((size_t)P * NF_K * 2);
    __bf16* wt_csn  = (__bf16*)alloc((size_t)P * SEM_D * 2);
    __bf16* wt_ig   = (__bf16*)alloc((size_t)GATE * IMG_D * 2);
    __bf16* wt_sg   = (__bf16*)alloc((size_t)GATE * SEM_D * 2);
    __bf16* wt_fg   = (__bf16*)alloc((size_t)GATE * IN_D * 2);
    __bf16* wt_gate = (__bf16*)alloc((size_t)CAT * CAT * 2);
    __bf16* wt_out  = (__bf16*)alloc((size_t)OUT_D * CAT * 2);
    __bf16* img_bf  = (__bf16*)alloc((size_t)Bb * NI * IMG_D * 2);
    __bf16* sem_bf  = (__bf16*)alloc((size_t)Bb * Ss * SEM_D * 2);
    float*  np_img  = (float*) alloc((size_t)ROWS * P * 4);
    float*  ipf     = (float*) alloc((size_t)Bb * NI * P * 4);
    float*  np_sem  = (float*) alloc((size_t)ROWS * P * 4);
    float*  spf     = (float*) alloc((size_t)Bb * Ss * P * 4);
    __bf16* img_ctx = (__bf16*)alloc((size_t)ROWS * IMG_D * 2);
    __bf16* sem_ctx = (__bf16*)alloc((size_t)ROWS * SEM_D * 2);
    __bf16* cat     = (__bf16*)alloc((size_t)ROWS * CAT * 2);
    __bf16* gc      = (__bf16*)alloc((size_t)ROWS * CAT * 2);

    dim3 b256(256);

    // --- precision conversion / layout prep ---
    k_node_feat<<<(ROWS * NF_K) / 256, b256, 0, stream>>>(h, que, nf);
    k_trans_bf16<<<dim3(P / 32,     NF_K / 32), b256, 0, stream>>>(W_cif,  wt_cif,  NF_K,  P);
    k_trans_bf16<<<dim3(P / 32,     IMG_D / 32), b256, 0, stream>>>(W_cii,  wt_cii,  IMG_D, P);
    k_trans_bf16<<<dim3(P / 32,     NF_K / 32), b256, 0, stream>>>(W_csf,  wt_csf,  NF_K,  P);
    k_trans_bf16<<<dim3(P / 32,     SEM_D / 32), b256, 0, stream>>>(W_csn,  wt_csn,  SEM_D, P);
    k_trans_bf16<<<dim3(GATE / 32,  IMG_D / 32), b256, 0, stream>>>(W_ig,   wt_ig,   IMG_D, GATE);
    k_trans_bf16<<<dim3(GATE / 32,  SEM_D / 32), b256, 0, stream>>>(W_sg,   wt_sg,   SEM_D, GATE);
    k_trans_bf16<<<dim3(GATE / 32,  IN_D / 32),  b256, 0, stream>>>(W_fg,   wt_fg,   IN_D,  GATE);
    k_trans_bf16<<<dim3(CAT / 32,   CAT / 32),   b256, 0, stream>>>(W_gate, wt_gate, CAT,   CAT);
    k_trans_bf16<<<dim3(OUT_D / 32, CAT / 32),   b256, 0, stream>>>(W_out,  wt_out,  CAT,   OUT_D);
    k_cvt_bf16<<<(Bb * NI * IMG_D) / 256, b256, 0, stream>>>(img, img_bf, Bb * NI * IMG_D);
    k_cvt_bf16<<<(Bb * Ss * SEM_D) / 256, b256, 0, stream>>>(sem, sem_bf, Bb * Ss * SEM_D);

    // --- cross-attention projections (WMMA GEMMs) ---
    launch_gemm(0, nf,     NF_K,  wt_cif, NF_K,  b_cif, np_img, nullptr, P, nullptr, ROWS,    P, NF_K,  stream);
    launch_gemm(0, img_bf, IMG_D, wt_cii, IMG_D, b_cii, ipf,    nullptr, P, nullptr, Bb * NI, P, IMG_D, stream);
    launch_gemm(0, nf,     NF_K,  wt_csf, NF_K,  b_csf, np_sem, nullptr, P, nullptr, ROWS,    P, NF_K,  stream);
    launch_gemm(0, sem_bf, SEM_D, wt_csn, SEM_D, b_csn, spf,    nullptr, P, nullptr, Bb * Ss, P, SEM_D, stream);

    // --- additive attention + context ---
    k_attn_img<<<ROWS, b256, 0, stream>>>(np_img, ipf, w_ia, b_ia, img, img_ctx);
    k_attn_sem<<<ROWS, b256, 0, stream>>>(np_sem, spf, w_sa, b_sa, sem, sem_ctx);

    // --- fusion projections into cat = [f | i | s]  (bf16 epilogue) ---
    launch_gemm(1, nf,      NF_K,  wt_fg, IN_D,  b_fg, nullptr, cat,          CAT, nullptr, ROWS, GATE, IN_D,  stream);
    launch_gemm(1, img_ctx, IMG_D, wt_ig, IMG_D, b_ig, nullptr, cat + GATE,   CAT, nullptr, ROWS, GATE, IMG_D, stream);
    launch_gemm(1, sem_ctx, SEM_D, wt_sg, SEM_D, b_sg, nullptr, cat + 2*GATE, CAT, nullptr, ROWS, GATE, SEM_D, stream);

    // --- gated fusion: gc = sigmoid(cat@W_gate + b) * cat  (fused epilogue) ---
    launch_gemm(2, cat, CAT, wt_gate, CAT, b_gate, nullptr, gc, CAT, cat, ROWS, CAT, CAT, stream);

    // --- output projection (f32 store to d_out) ---
    launch_gemm(0, gc, CAT, wt_out, CAT, b_out, out, nullptr, OUT_D, nullptr, ROWS, OUT_D, CAT, stream);

    (void)in_sizes; (void)n_in; (void)out_size; (void)ws_size;
}
